// MultiheadAttention_11252814315867
// MI455X (gfx1250) — matrix operand order repaired and checked by reading, not compiled, not run on hardware
//
#include <hip/hip_runtime.h>
#include <hip/hip_bf16.h>

typedef __attribute__((ext_vector_type(16))) _Float16 v16h;
typedef __attribute__((ext_vector_type(8)))  _Float16 v8h;
typedef __attribute__((ext_vector_type(8)))  float    v8f;

#define EPSF      1e-5f
#define MASKFILL  -50000.0f

constexpr int Bn  = 8;
constexpr int Sn  = 1024;
constexpr int Dn  = 512;
constexpr int Hn  = 8;
constexpr int HDn = 64;           // head dim
constexpr int SB  = Bn * Sn;      // 8192 total rows
constexpr int WPB = 8;            // waves per 256-thread block (wave32)

// ---------------------------------------------------------------------------
// WMMA fragment coordinates as this code loads them (contiguous k order, wave32)
//   A (16x32 f16): lanes 0-15 -> M=lane, K = 0..15;         lanes 16-31 -> K=16..31
//   B (32x16 f16): lanes 0-15 -> N=lane, K = 0..15;         lanes 16-31 -> K=16..31
//   C/D (16x16 f32): reg r -> M = r + (lane<16?0:8), N = lane&15
// ---------------------------------------------------------------------------

// --- CDNA5 async global->LDS copy (GLOBAL_LOAD_ASYNC_TO_LDS_B128, ASYNCcnt) ---
// IOFFSET is added to BOTH the LDS and the global address (ISA 15.18.3), so a
// 32B row is two b128 copies off one base pair.
__device__ __forceinline__ void async_copy_row32B(uint32_t lds_off, uint64_t gaddr)
{
    asm volatile("global_load_async_to_lds_b128 %0, %1, off\n\t"
                 "global_load_async_to_lds_b128 %0, %1, off offset:16"
                 :: "v"(lds_off), "v"(gaddr) : "memory");
}
__device__ __forceinline__ void async_wait_all()
{
    asm volatile("s_wait_asynccnt 0x0" ::: "memory");
}

// ========================= masked set-norm statistics ======================
__global__ __launch_bounds__(256)
void row_stats_kernel(const float* __restrict__ Q, float* __restrict__ fmask,
                      float* __restrict__ red)
{
    const int row = blockIdx.x;          // one block per (b,s) row
    const int tid = threadIdx.x;
    const float* p = Q + (size_t)row * Dn;
    float s = 0.f, ss = 0.f; int nz = 0;
    for (int j = tid; j < Dn; j += 256) {
        float x = p[j];
        s += x; ss += x * x; nz |= (x != 0.f);
    }
    __shared__ float s1[256], s2[256];
    __shared__ int   s3[256];
    s1[tid] = s; s2[tid] = ss; s3[tid] = nz;
    __syncthreads();
    for (int st = 128; st > 0; st >>= 1) {
        if (tid < st) { s1[tid] += s1[tid+st]; s2[tid] += s2[tid+st]; s3[tid] |= s3[tid+st]; }
        __syncthreads();
    }
    if (tid == 0) {
        const int b = row >> 10;                     // row / Sn
        fmask[row] = s3[0] ? 1.f : 0.f;
        if (s3[0]) {
            atomicAdd(&red[3*b + 0], s1[0]);
            atomicAdd(&red[3*b + 1], s2[0]);
            atomicAdd(&red[3*b + 2], (float)Dn);     // cnt = nvalid * D
        }
    }
}

__global__ __launch_bounds__(32)
void finalize_stats_kernel(const float* __restrict__ red, float* __restrict__ mv)
{
    const int b = threadIdx.x;
    if (b < Bn) {
        const float cnt  = red[3*b + 2];
        const float mean = (cnt > 0.f) ? red[3*b + 0] / cnt : 0.f;
        const float var  = (cnt > 0.f) ? red[3*b + 1] / cnt - mean * mean : 0.f;
        mv[2*b + 0] = mean;
        mv[2*b + 1] = rsqrtf(var + EPSF);
    }
}

// ============ normalize Q -> f16 (Qn = (Q-mean)*rstd*fmask) ================
__global__ __launch_bounds__(256)
void normalize_f16_kernel(const float* __restrict__ Q, const float* __restrict__ fmask,
                          const float* __restrict__ mv, _Float16* __restrict__ Qn)
{
    const size_t idx = (size_t)blockIdx.x * 256 + threadIdx.x;   // SB*Dn launched exactly
    const int row = (int)(idx / Dn);
    const int b   = row >> 10;
    const float x = (Q[idx] - mv[2*b]) * mv[2*b + 1] * fmask[row];
    Qn[idx] = (_Float16)x;
}

__global__ __launch_bounds__(256)
void cvt_f16_kernel(const float* __restrict__ src, _Float16* __restrict__ dst)
{
    const size_t idx = (size_t)blockIdx.x * 256 + threadIdx.x;   // Dn*Dn launched exactly
    dst[idx] = (_Float16)src[idx];
}

// ============ projection GEMM: Out = (A @ W + bias) * fmask  (f16) =========
// Block owns one 16-wide output column panel; the 512x16 W panel (16 KB) is
// staged in LDS via GLOBAL_LOAD_ASYNC_TO_LDS_B128 and shared by 8 waves that
// each compute one 16x16 tile (128 rows per block).
__global__ __launch_bounds__(256)
void proj_gemm_kernel(const _Float16* __restrict__ A, const _Float16* __restrict__ W,
                      const float* __restrict__ bias, const float* __restrict__ fmask,
                      _Float16* __restrict__ Out)
{
    __shared__ _Float16 wp[Dn * 16];                 // [k][c] panel, 16 KB
    const int tid  = threadIdx.x;
    const int lane = tid & 31;
    const int wv   = tid >> 5;

    const int tn = blockIdx.x % (Dn / 16);
    const int tg = blockIdx.x / (Dn / 16);
    const int tm = tg * WPB + wv;                    // this wave's row tile

    // ---- async stage of W panel: thread t copies rows 2t, 2t+1 (32B each) ----
    {
        const int r0 = 2 * tid;
        async_copy_row32B((uint32_t)(uintptr_t)(wp + (size_t)r0 * 16),
                          (uint64_t)(uintptr_t)(W + (size_t)r0 * Dn + tn * 16));
        async_copy_row32B((uint32_t)(uintptr_t)(wp + (size_t)(r0 + 1) * 16),
                          (uint64_t)(uintptr_t)(W + (size_t)(r0 + 1) * Dn + tn * 16));
        async_wait_all();
    }
    __syncthreads();

    const int m   = lane & 15;
    const int kb  = (lane & 16) ? 16 : 0;    // A K-half select
    const int n   = lane & 15;
    const int kb2 = (lane & 16) ? 16 : 0;    // B K-half select
    const int arow = tm * 16 + m;

    v8f acc = {};
    for (int k0 = 0; k0 < Dn; k0 += 32) {
        const _Float16* pa = A + (size_t)arow * Dn + k0 + kb;
        v16h a;
        #pragma unroll
        for (int i = 0; i < 16; ++i) a[i] = pa[i];

        const _Float16* pb = wp + (size_t)(k0 + kb2) * 16 + n;   // LDS, stride 16 halves
        v16h bf;
        #pragma unroll
        for (int i = 0; i < 16; ++i) bf[i] = pb[i * 16];

        acc = __builtin_amdgcn_wmma_f32_16x16x32_f16(false, a, false, bf,
                                                     (short)0, acc, false, false);
    }
    const int col = tn * 16 + n;
    #pragma unroll
    for (int r = 0; r < 8; ++r) {
        const int mr   = ((lane & 16) ? 8 : 0) + r;
        const int orow = tm * 16 + mr;
        float v = acc[r] + bias[col];
        v *= fmask[orow];
        Out[(size_t)orow * Dn + col] = (_Float16)v;
    }
}

// ===== fused attention strip: scores(WMMA) -> softmax(LDS) -> w out -> w@V =====
// One block per (b, h, 16-query strip). 16x1024 fp32 score strip lives in LDS (64KB).
__global__ __launch_bounds__(256)
void attn_fused_kernel(const _Float16* __restrict__ Qp, const _Float16* __restrict__ Kp,
                       const _Float16* __restrict__ Vp, const float* __restrict__ fmask,
                       float* __restrict__ Wout, float* __restrict__ attn)
{
    __shared__ float sc[16 * Sn];                    // 64 KB strip (reused for partials)
    const int lane = threadIdx.x & 31;
    const int wv   = threadIdx.x >> 5;               // wave id 0..7

    int blk = blockIdx.x;
    const int qt = blk % (Sn / 16);  blk /= (Sn / 16);
    const int h  = blk % Hn;
    const int b  = blk / Hn;

    const int m   = lane & 15;
    const int kb  = (lane & 16) ? 16 : 0;
    const int n   = lane & 15;
    const int kb2 = (lane & 16) ? 16 : 0;

    // ---- A fragments for Q strip (16 x 64), invariant across key tiles ----
    v16h a0, a1;
    {
        const int qrow = b * Sn + qt * 16 + m;
        const _Float16* pa = Qp + (size_t)qrow * Dn + h * HDn + kb;
        #pragma unroll
        for (int i = 0; i < 16; ++i) a0[i] = pa[i];
        pa += 32;
        #pragma unroll
        for (int i = 0; i < 16; ++i) a1[i] = pa[i];
    }

    // ---- scores: each wave computes 8 of the 64 key tiles -> LDS ----
    for (int kt = wv * 8; kt < wv * 8 + 8; ++kt) {
        const int krow = b * Sn + kt * 16 + n;
        const _Float16* pb = Kp + (size_t)krow * Dn + h * HDn + kb2;
        v16h b0, b1;
        #pragma unroll
        for (int i = 0; i < 16; ++i) b0[i] = pb[i];
        #pragma unroll
        for (int i = 0; i < 16; ++i) b1[i] = pb[32 + i];

        v8f acc = {};
        acc = __builtin_amdgcn_wmma_f32_16x16x32_f16(false, a0, false, b0,
                                                     (short)0, acc, false, false);
        acc = __builtin_amdgcn_wmma_f32_16x16x32_f16(false, a1, false, b1,
                                                     (short)0, acc, false, false);
        const float mk = fmask[krow];
        #pragma unroll
        for (int r = 0; r < 8; ++r) {
            const int mr = ((lane & 16) ? 8 : 0) + r;
            const float mq = fmask[b * Sn + qt * 16 + mr];
            sc[mr * Sn + kt * 16 + n] =
                acc[r] * 0.125f + ((mq * mk != 0.f) ? 0.f : MASKFILL);
        }
    }
    __syncthreads();

    // ---- softmax in place: wave wv handles rows 2wv (lanes 0-15) / 2wv+1 ----
    {
        const int row = 2 * wv + ((lane >> 4) & 1);
        float* p = sc + row * Sn;
        const int c0 = (lane & 15) * 64;             // 64 columns per lane
        float mx = -3.4e38f;
        for (int j = 0; j < 64; ++j) mx = fmaxf(mx, p[c0 + j]);
        #pragma unroll
        for (int s = 8; s > 0; s >>= 1) mx = fmaxf(mx, __shfl_xor(mx, s, 32));
        float sum = 0.f;
        for (int j = 0; j < 64; ++j) sum += __expf(p[c0 + j] - mx);
        #pragma unroll
        for (int s = 8; s > 0; s >>= 1) sum += __shfl_xor(sum, s, 32);
        const float inv = 1.0f / sum;
        const float mq = fmask[b * Sn + qt * 16 + row];
        for (int j = 0; j < 64; ++j) {
            const int c = c0 + j;
            p[c] = __expf(p[c] - mx) * inv * mq * fmask[b * Sn + c];
        }
    }
    __syncthreads();

    // ---- single coalesced store of the w strip (the dominant HBM traffic) ----
    {
        float* wout = Wout + ((size_t)(b * Hn + h) * Sn + qt * 16) * Sn;
        for (int idx = threadIdx.x; idx < 16 * Sn; idx += 256)
            wout[idx] = sc[idx];
    }

    // ---- attn strip = w(16x1024) @ V(1024x64); wave -> (nt = wv&3, khalf) ----
    {
        const int nt = wv & 3, khalf = wv >> 2;
        v8f acc = {};
        for (int k0 = khalf * 512; k0 < khalf * 512 + 512; k0 += 32) {
            const float* pa = sc + m * Sn + k0 + kb;
            v16h aa;
            #pragma unroll
            for (int i = 0; i < 16; ++i) {
                aa[i]     = (_Float16)pa[i];
            }
            const _Float16* pb = Vp + (size_t)(b * Sn + k0 + kb2) * Dn + h * HDn + nt * 16 + n;
            v16h bb;
            #pragma unroll
            for (int i = 0; i < 16; ++i) bb[i] = pb[(size_t)i * Dn];
            acc = __builtin_amdgcn_wmma_f32_16x16x32_f16(false, aa, false, bb,
                                                         (short)0, acc, false, false);
        }
        __syncthreads();                  // all strip reads done -> reuse LDS for partials
        float* part = sc + wv * 256;      // 16x16 f32 partial per wave
        #pragma unroll
        for (int r = 0; r < 8; ++r) {
            const int mr = ((lane & 16) ? 8 : 0) + r;
            part[mr * 16 + n] = acc[r];
        }
    }
    __syncthreads();

    // ---- combine the two K-half partials and write attn (pre-residual) ----
    for (int idx = threadIdx.x; idx < 16 * HDn; idx += 256) {
        const int mr = idx >> 6;          // 0..15 query within strip
        const int c  = idx & 63;          // 0..63 head-dim column
        const int nt = c >> 4, cn = c & 15;
        const float v = sc[nt * 256 + mr * 16 + cn] + sc[(4 + nt) * 256 + mr * 16 + cn];
        attn[(size_t)(b * Sn + qt * 16 + mr) * Dn + h * HDn + c] = v;
    }
}

// ======== residual add (attn += Q) + masked stats for second set_norm ======
__global__ __launch_bounds__(256)
void add_q_stats_kernel(const float* __restrict__ Q, float* __restrict__ attn,
                        const float* __restrict__ fmask, float* __restrict__ red)
{
    const int row = blockIdx.x;
    const int tid = threadIdx.x;
    const float fm = fmask[row];
    float s = 0.f, ss = 0.f;
    for (int j = tid; j < Dn; j += 256) {
        const size_t idx = (size_t)row * Dn + j;
        const float x = attn[idx] + Q[idx];
        attn[idx] = x;
        s += x * fm; ss += x * x * fm;
    }
    __shared__ float s1[256], s2[256];
    s1[tid] = s; s2[tid] = ss;
    __syncthreads();
    for (int st = 128; st > 0; st >>= 1) {
        if (tid < st) { s1[tid] += s1[tid+st]; s2[tid] += s2[tid+st]; }
        __syncthreads();
    }
    if (tid == 0 && fm > 0.f) {
        const int b = row >> 10;
        atomicAdd(&red[3*b + 0], s1[0]);
        atomicAdd(&red[3*b + 1], s2[0]);
        atomicAdd(&red[3*b + 2], (float)Dn);
    }
}

// ===== final: out = attn + relu(set_norm(attn) @ Wo + bo), norm on the fly =====
// Same LDS panel staging as proj_gemm; A is normalized on the fly from fp32.
__global__ __launch_bounds__(256)
void out_gemm_kernel(const float* __restrict__ attn, const _Float16* __restrict__ Wo,
                     const float* __restrict__ bo, const float* __restrict__ fmask,
                     const float* __restrict__ mv, float* __restrict__ Out)
{
    __shared__ _Float16 wp[Dn * 16];                 // [k][c] panel, 16 KB
    const int tid  = threadIdx.x;
    const int lane = tid & 31;
    const int wv   = tid >> 5;

    const int tn = blockIdx.x % (Dn / 16);
    const int tg = blockIdx.x / (Dn / 16);
    const int tm = tg * WPB + wv;

    {
        const int r0 = 2 * tid;
        async_copy_row32B((uint32_t)(uintptr_t)(wp + (size_t)r0 * 16),
                          (uint64_t)(uintptr_t)(Wo + (size_t)r0 * Dn + tn * 16));
        async_copy_row32B((uint32_t)(uintptr_t)(wp + (size_t)(r0 + 1) * 16),
                          (uint64_t)(uintptr_t)(Wo + (size_t)(r0 + 1) * Dn + tn * 16));
        async_wait_all();
    }
    __syncthreads();

    const int m   = lane & 15;
    const int kb  = (lane & 16) ? 16 : 0;
    const int n   = lane & 15;
    const int kb2 = (lane & 16) ? 16 : 0;

    const int arow  = tm * 16 + m;
    const int batch = arow >> 10;
    const float mean = mv[2*batch], inv = mv[2*batch + 1];
    const float fm   = fmask[arow];

    v8f acc = {};
    for (int k0 = 0; k0 < Dn; k0 += 32) {
        const float* pa = attn + (size_t)arow * Dn + k0 + kb;
        v16h a;
        #pragma unroll
        for (int i = 0; i < 16; ++i) {
            a[i]     = (_Float16)((pa[i]      - mean) * inv * fm);
        }
        const _Float16* pb = wp + (size_t)(k0 + kb2) * 16 + n;
        v16h bf;
        #pragma unroll
        for (int i = 0; i < 16; ++i) bf[i] = pb[i * 16];
        acc = __builtin_amdgcn_wmma_f32_16x16x32_f16(false, a, false, bf,
                                                     (short)0, acc, false, false);
    }
    const int col = tn * 16 + n;
    #pragma unroll
    for (int r = 0; r < 8; ++r) {
        const int mr   = ((lane & 16) ? 8 : 0) + r;
        const int orow = tm * 16 + mr;
        const float v  = fmaxf(acc[r] + bo[col], 0.f);
        Out[(size_t)orow * Dn + col] = attn[(size_t)orow * Dn + col] + v;
    }
}

// ===========================================================================
extern "C" void kernel_launch(void* const* d_in, const int* in_sizes, int n_in,
                              void* d_out, int out_size, void* d_ws, size_t ws_size,
                              hipStream_t stream)
{
    (void)in_sizes; (void)n_in; (void)out_size; (void)ws_size;

    const float* Q  = (const float*)d_in[0];
    const float* Wq = (const float*)d_in[1];
    const float* bq = (const float*)d_in[2];
    const float* Wk = (const float*)d_in[3];
    const float* bk = (const float*)d_in[4];
    const float* Wv = (const float*)d_in[5];
    const float* bv = (const float*)d_in[6];
    const float* Wo = (const float*)d_in[7];
    const float* bo = (const float*)d_in[8];

    float* out_final = (float*)d_out;                       // B*S*D fp32
    float* w_attn    = out_final + (size_t)SB * Dn;         // B*H*S*S fp32

    // ---- carve workspace (~50 MiB) ----
    char* base = (char*)d_ws;
    size_t off = 0;
    auto carve = [&](size_t bytes) -> char* {
        char* p = base + off;
        off = (off + bytes + 255) & ~(size_t)255;
        return p;
    };
    float*    fmask = (float*)   carve((size_t)SB * sizeof(float));
    float*    red1  = (float*)   carve(3 * Bn * sizeof(float));
    float*    mv1   = (float*)   carve(2 * Bn * sizeof(float));
    float*    red2  = (float*)   carve(3 * Bn * sizeof(float));
    float*    mv2   = (float*)   carve(2 * Bn * sizeof(float));
    _Float16* Qn16  = (_Float16*)carve((size_t)SB * Dn * 2);
    _Float16* Wq16  = (_Float16*)carve((size_t)Dn * Dn * 2);
    _Float16* Wk16  = (_Float16*)carve((size_t)Dn * Dn * 2);
    _Float16* Wv16  = (_Float16*)carve((size_t)Dn * Dn * 2);
    _Float16* Wo16  = (_Float16*)carve((size_t)Dn * Dn * 2);
    _Float16* Qp16  = (_Float16*)carve((size_t)SB * Dn * 2);
    _Float16* Kp16  = (_Float16*)carve((size_t)SB * Dn * 2);
    _Float16* Vp16  = (_Float16*)carve((size_t)SB * Dn * 2);
    float*    attn  = (float*)   carve((size_t)SB * Dn * sizeof(float));

    hipMemsetAsync(red1, 0, 3 * Bn * sizeof(float), stream);
    hipMemsetAsync(red2, 0, 3 * Bn * sizeof(float), stream);

    // 1) masked mean/var stats + row mask
    row_stats_kernel<<<SB, 256, 0, stream>>>(Q, fmask, red1);
    finalize_stats_kernel<<<1, 32, 0, stream>>>(red1, mv1);

    // 2) set_norm(Q) -> f16, weights -> f16
    normalize_f16_kernel<<<(SB * Dn) / 256, 256, 0, stream>>>(Q, fmask, mv1, Qn16);
    cvt_f16_kernel<<<(Dn * Dn) / 256, 256, 0, stream>>>(Wq, Wq16);
    cvt_f16_kernel<<<(Dn * Dn) / 256, 256, 0, stream>>>(Wk, Wk16);
    cvt_f16_kernel<<<(Dn * Dn) / 256, 256, 0, stream>>>(Wv, Wv16);
    cvt_f16_kernel<<<(Dn * Dn) / 256, 256, 0, stream>>>(Wo, Wo16);

    // 3) Q/K/V projections (WMMA + async LDS weight-panel staging)
    const int proj_blocks = (SB / 16 / WPB) * (Dn / 16);    // 2048
    proj_gemm_kernel<<<proj_blocks, 256, 0, stream>>>(Qn16, Wq16, bq, fmask, Qp16);
    proj_gemm_kernel<<<proj_blocks, 256, 0, stream>>>(Qn16, Wk16, bk, fmask, Kp16);
    proj_gemm_kernel<<<proj_blocks, 256, 0, stream>>>(Qn16, Wv16, bv, fmask, Vp16);

    // 4) fused attention: scores -> softmax -> w (single HBM store) -> w@V
    attn_fused_kernel<<<Bn * Hn * (Sn / 16), 256, 0, stream>>>(
        Qp16, Kp16, Vp16, fmask, w_attn, attn);

    // 5) residual + second set_norm stats
    add_q_stats_kernel<<<SB, 256, 0, stream>>>(Q, attn, fmask, red2);
    finalize_stats_kernel<<<1, 32, 0, stream>>>(red2, mv2);

    // 6) final projection with on-the-fly set_norm, ReLU, residual
    out_gemm_kernel<<<proj_blocks, 256, 0, stream>>>(attn, Wo16, bo, fmask, mv2, out_final);
}
